// CLIPVisionTransformer_70179765616850
// MI455X (gfx1250) — matrix operand order repaired and checked by reading, not compiled, not run on hardware
//
#include <hip/hip_runtime.h>
#include <hip/hip_bf16.h>

// ---------------------------------------------------------------------------
// CLIP ViT forward for gfx1250 (MI455X): all GEMMs via v_wmma_f32_16x16x32_bf16
// ---------------------------------------------------------------------------

typedef __attribute__((ext_vector_type(16))) __bf16 v16bf;
typedef __attribute__((ext_vector_type(8)))  __bf16 v8bf;
typedef __attribute__((ext_vector_type(8)))  float  v8f;

#define DEPTH_ 12
#define B_   32
#define NTOK 197
#define NT_  (B_ * NTOK)      // 6304
#define D_   768
#define H_   12
#define DH_  64
#define MLP_ 3072
#define NP_  196
#define MPATCH (B_ * NP_)     // 6272
#define NPAD 208              // 197 padded to mult of 16 (M side)
#define KPAD 224              // 197 padded to mult of 32 (K side)
#define BH_  (B_ * H_)        // 384
#define PROJ_ 512
#define NCLS 1000
#define NCLSP 1008

// ---------------------------------------------------------------------------
// Generic batched GEMM:  C = alpha * (A @ W^T) + bias + add
//   A:  [M, K]  bf16 row-major (lda), batch stride sA
//   W:  [N, K]  bf16 row-major (ldw), batch stride sW  (i.e. B[k][n] = W[n][k])
//   C:  [M, N]  f32 row-major (ldc), batch stride sC; stores masked to n<Nreal
// One wave computes a 16(M) x 64(N) tile; block = 4 waves stacked in M.
// Requires: M % 16 == 0, K % 32 == 0, Nfull % 16 == 0, Nfull >= 64.
// Inner loop is branchless: out-of-range N tiles are clamped for loads and
// masked only at the store epilogue, so accumulators never cross control flow.
// B pointers kept as individual scalar SSA values (not an array) so the
// addrspace-inference pass emits global_load_b128 (not flat_load_b128).
// ---------------------------------------------------------------------------
__global__ void __launch_bounds__(128)
gemm_bf16_kernel(const __bf16* __restrict__ A, int lda, long long sA,
                 const __bf16* __restrict__ W, int ldw, long long sW,
                 const float* __restrict__ bias,
                 const float* __restrict__ add,
                 float* __restrict__ C, int ldc, long long sC,
                 int M, int Nfull, int Nreal, int K, float alpha)
{
    const int lane = threadIdx.x & 31;
    const int wave = threadIdx.x >> 5;
    const int m0 = (blockIdx.y * 4 + wave) * 16;
    if (m0 >= M) return;                       // wave-uniform
    const int n0 = blockIdx.x * 64;

    const long long batch = blockIdx.z;
    A += batch * sA;
    W += batch * sW;
    C += batch * sC;
    const float* addp = add ? (add + batch * sC) : nullptr;

    const int half = lane >> 4;                // 0: lanes 0-15, 1: lanes 16-31
    const int r    = lane & 15;

    // Per-tile column bases; clamp so loads stay in-bounds even for tiles past
    // Nfull (their results are discarded at store time).  Wave-uniform.
    int nb0 = n0;
    int nb1 = n0 + 16;
    int nb2 = n0 + 32;
    int nb3 = n0 + 48;
    const int nmax = Nfull - 16;
    if (nb1 > nmax) nb1 = nmax;
    if (nb2 > nmax) nb2 = nmax;
    if (nb3 > nmax) nb3 = nmax;

    const __bf16* wp0 = W + (long long)(nb0 + r) * ldw + 16 * half;
    const __bf16* wp1 = W + (long long)(nb1 + r) * ldw + 16 * half;
    const __bf16* wp2 = W + (long long)(nb2 + r) * ldw + 16 * half;
    const __bf16* wp3 = W + (long long)(nb3 + r) * ldw + 16 * half;

    // A row base for this lane: row m0+r, starting at K offset 16*half
    const __bf16* ap = A + (long long)(m0 + r) * lda + 16 * half;

    v8f acc0 = {}, acc1 = {}, acc2 = {}, acc3 = {};

    for (int k = 0; k < K; k += 32) {
        // A 16x32 bf16 fragment (contiguous K, matching B):
        //  lanes 0-15 : K = k+0..15
        //  lanes 16-31: K = k+16..31
        v8bf alo = *(const v8bf*)ap;
        v8bf ahi = *(const v8bf*)(ap + 8);
        ap += 32;
        v16bf afrag = __builtin_shufflevector(alo, ahi,
            0,1,2,3,4,5,6,7,8,9,10,11,12,13,14,15);

        // B 32x16 fragments: lane holds column n, K range k+16*half..+15,
        // contiguous in W row n -> single 32-byte load each.
        v16bf b0 = *(const v16bf*)wp0;  wp0 += 32;
        v16bf b1 = *(const v16bf*)wp1;  wp1 += 32;
        v16bf b2 = *(const v16bf*)wp2;  wp2 += 32;
        v16bf b3 = *(const v16bf*)wp3;  wp3 += 32;

        acc0 = __builtin_amdgcn_wmma_f32_16x16x32_bf16(
            false, afrag, false, b0, (short)0, acc0, false, false);
        acc1 = __builtin_amdgcn_wmma_f32_16x16x32_bf16(
            false, afrag, false, b1, (short)0, acc1, false, false);
        acc2 = __builtin_amdgcn_wmma_f32_16x16x32_bf16(
            false, afrag, false, b2, (short)0, acc2, false, false);
        acc3 = __builtin_amdgcn_wmma_f32_16x16x32_bf16(
            false, afrag, false, b3, (short)0, acc3, false, false);
    }

    // Epilogue: D layout — VGPR vr holds M = m0 + vr + 8*half, N = n0+16t+r
    v8f accs[4] = {acc0, acc1, acc2, acc3};
#pragma unroll
    for (int t = 0; t < 4; ++t) {
        if (n0 + 16 * t >= Nfull) continue;    // wave-uniform
        const int col = n0 + 16 * t + r;
        if (col >= Nreal) continue;
        const float binit = bias ? bias[col] : 0.0f;
#pragma unroll
        for (int vr = 0; vr < 8; ++vr) {
            const int row = m0 + vr + 8 * half;
            const long long ci = (long long)row * ldc + col;
            float v = alpha * accs[t][vr] + binit;
            if (addp) v += addp[ci];
            C[ci] = v;
        }
    }
}

// ---------------------------------------------------------------------------
// Elementwise / reshape kernels
// ---------------------------------------------------------------------------
__global__ void cvt_f32_bf16_kernel(const float* __restrict__ src,
                                    __bf16* __restrict__ dst, long long n)
{
    long long i = (long long)blockIdx.x * blockDim.x + threadIdx.x;
    long long stride = (long long)gridDim.x * blockDim.x;
    for (; i < n; i += stride) dst[i] = (__bf16)src[i];
}

// Pad rows: dst [rowspad, cols], src [rows, cols]; rows beyond -> 0
__global__ void cvt_pad_rows_kernel(const float* __restrict__ src,
                                    __bf16* __restrict__ dst,
                                    int rows, int cols, long long n)
{
    long long i = (long long)blockIdx.x * blockDim.x + threadIdx.x;
    long long stride = (long long)gridDim.x * blockDim.x;
    for (; i < n; i += stride) {
        long long row = i / cols;
        dst[i] = (row < rows) ? (__bf16)src[i] : (__bf16)0.0f;
    }
}

// im2col of NCHW input into bf16 patch matrix [B*196, 768]
__global__ void im2col_kernel(const float* __restrict__ x,
                              __bf16* __restrict__ patches, long long n)
{
    long long i = (long long)blockIdx.x * blockDim.x + threadIdx.x;
    long long stride = (long long)gridDim.x * blockDim.x;
    for (; i < n; i += stride) {
        int k  = (int)(i % 768);
        long long bp = i / 768;
        int b  = (int)(bp / NP_);
        int p  = (int)(bp % NP_);
        int ph = p / 14, pw = p % 14;
        int c  = k / 256;
        int rem = k % 256;
        int py = rem / 16, px = rem % 16;
        long long src = (((long long)b * 3 + c) * 224 + (ph * 16 + py)) * 224
                        + (pw * 16 + px);
        patches[i] = (__bf16)x[src];
    }
}

// Build token stream: y[b,0,:] = cls + pos[0]; y[b,1+p,:] = patch_out + pos
__global__ void assemble_kernel(const float* __restrict__ patch_out,
                                const float* __restrict__ cls_token,
                                const float* __restrict__ pos_embed,
                                float* __restrict__ y, long long n)
{
    long long i = (long long)blockIdx.x * blockDim.x + threadIdx.x;
    long long stride = (long long)gridDim.x * blockDim.x;
    for (; i < n; i += stride) {
        int d = (int)(i % D_);
        long long row = i / D_;
        int b = (int)(row / NTOK);
        int t = (int)(row % NTOK);
        float v = (t == 0) ? cls_token[d]
                           : patch_out[((long long)b * NP_ + (t - 1)) * D_ + d];
        y[i] = v + pos_embed[(long long)t * D_ + d];
    }
}

// Row LayerNorm. in row stride configurable (for strided cls extraction).
__global__ void ln_kernel(const float* __restrict__ in, long long row_stride,
                          int D, const float* __restrict__ w,
                          const float* __restrict__ bvec,
                          float* __restrict__ outf, __bf16* __restrict__ outb)
{
    const int row = blockIdx.x;
    const float* x = in + (long long)row * row_stride;
    __shared__ float red[256];
    const int tid = threadIdx.x;

    float s = 0.0f;
    for (int i = tid; i < D; i += blockDim.x) s += x[i];
    red[tid] = s; __syncthreads();
    for (int off = 128; off > 0; off >>= 1) {
        if (tid < off) red[tid] += red[tid + off];
        __syncthreads();
    }
    const float mean = red[0] / D;
    __syncthreads();

    float v = 0.0f;
    for (int i = tid; i < D; i += blockDim.x) { float d = x[i] - mean; v += d * d; }
    red[tid] = v; __syncthreads();
    for (int off = 128; off > 0; off >>= 1) {
        if (tid < off) red[tid] += red[tid + off];
        __syncthreads();
    }
    const float rstd = rsqrtf(red[0] / D + 1e-5f);

    for (int i = tid; i < D; i += blockDim.x) {
        float y = (x[i] - mean) * rstd * w[i] + bvec[i];
        if (outf) outf[(long long)row * D + i] = y;
        if (outb) outb[(long long)row * D + i] = (__bf16)y;
    }
}

// Scatter qkv rows [B*197, 2304] into padded per-head Q/K layout
// dst[(bh*npad + t)*64 + d], zero for t >= 197.  sidx selects q(0)/k(1).
__global__ void fill_qk_kernel(const float* __restrict__ qkv,
                               __bf16* __restrict__ dst, int npad, int sidx,
                               long long n)
{
    long long i = (long long)blockIdx.x * blockDim.x + threadIdx.x;
    long long stride = (long long)gridDim.x * blockDim.x;
    for (; i < n; i += stride) {
        int d  = (int)(i % 64);
        int t  = (int)((i / 64) % npad);
        int bh = (int)(i / (64LL * npad));
        int b = bh / H_, h = bh % H_;
        float v = 0.0f;
        if (t < NTOK)
            v = qkv[((long long)b * NTOK + t) * (3 * D_) + sidx * D_ + h * 64 + d];
        dst[i] = (__bf16)v;
    }
}

// V transposed: vt[(bh*64 + d)*KPAD + j] = V[j, d], zero for j >= 197
__global__ void fill_vt_kernel(const float* __restrict__ qkv,
                               __bf16* __restrict__ dst, long long n)
{
    long long i = (long long)blockIdx.x * blockDim.x + threadIdx.x;
    long long stride = (long long)gridDim.x * blockDim.x;
    for (; i < n; i += stride) {
        int j  = (int)(i % KPAD);
        int d  = (int)((i / KPAD) % 64);
        int bh = (int)(i / (64LL * KPAD));
        int b = bh / H_, h = bh % H_;
        float v = 0.0f;
        if (j < NTOK)
            v = qkv[((long long)b * NTOK + j) * (3 * D_) + 2 * D_ + h * 64 + d];
        dst[i] = (__bf16)v;
    }
}

// Masked softmax over a [*, KPAD] row (valid cols < 197), write bf16
__global__ void softmax_kernel(const float* __restrict__ sc,
                               __bf16* __restrict__ at)
{
    const long long row = blockIdx.x;
    const float* s = sc + row * KPAD;
    __bf16* o = at + row * KPAD;
    __shared__ float red[256];
    const int tid = threadIdx.x;

    float m = -1e30f;
    for (int j = tid; j < NTOK; j += blockDim.x) m = fmaxf(m, s[j]);
    red[tid] = m; __syncthreads();
    for (int off = 128; off > 0; off >>= 1) {
        if (tid < off) red[tid] = fmaxf(red[tid], red[tid + off]);
        __syncthreads();
    }
    m = red[0];
    __syncthreads();

    float sum = 0.0f;
    for (int j = tid; j < NTOK; j += blockDim.x) sum += __expf(s[j] - m);
    red[tid] = sum; __syncthreads();
    for (int off = 128; off > 0; off >>= 1) {
        if (tid < off) red[tid] += red[tid + off];
        __syncthreads();
    }
    const float inv = 1.0f / red[0];

    for (int j = tid; j < KPAD; j += blockDim.x)
        o[j] = (j < NTOK) ? (__bf16)(__expf(s[j] - m) * inv) : (__bf16)0.0f;
}

// Gather per-head attn output [BH, 208, 64] -> token stream bf16 [6304, 768]
__global__ void gather_o_kernel(const float* __restrict__ oh,
                                __bf16* __restrict__ dst, long long n)
{
    long long i = (long long)blockIdx.x * blockDim.x + threadIdx.x;
    long long stride = (long long)gridDim.x * blockDim.x;
    for (; i < n; i += stride) {
        int d768 = (int)(i % D_);
        long long row = i / D_;
        int b = (int)(row / NTOK);
        int t = (int)(row % NTOK);
        int h = d768 / 64, d = d768 % 64;
        dst[i] = (__bf16)oh[(((long long)b * H_ + h) * NPAD + t) * 64 + d];
    }
}

// Exact GELU (erf form) f32 -> bf16
__global__ void gelu_kernel(const float* __restrict__ src,
                            __bf16* __restrict__ dst, long long n)
{
    long long i = (long long)blockIdx.x * blockDim.x + threadIdx.x;
    long long stride = (long long)gridDim.x * blockDim.x;
    for (; i < n; i += stride) {
        float x = src[i];
        dst[i] = (__bf16)(0.5f * x * (1.0f + erff(x * 0.7071067811865475f)));
    }
}

// ---------------------------------------------------------------------------
// Host launcher
// ---------------------------------------------------------------------------
static inline int cdiv_i(long long a, long long b) { return (int)((a + b - 1) / b); }

extern "C" void kernel_launch(void* const* d_in, const int* in_sizes, int n_in,
                              void* d_out, int out_size, void* d_ws, size_t ws_size,
                              hipStream_t stream)
{
    const float* x_in      = (const float*)d_in[0];
    const float* conv_w    = (const float*)d_in[1];
    const float* cls_token = (const float*)d_in[2];
    const float* pos_embed = (const float*)d_in[3];
    const float* ln_pre_w  = (const float*)d_in[4];
    const float* ln_pre_b  = (const float*)d_in[5];
    const float* ln1_w     = (const float*)d_in[6];
    const float* ln1_b     = (const float*)d_in[7];
    const float* qkv_w     = (const float*)d_in[8];
    const float* qkv_b     = (const float*)d_in[9];
    const float* aproj_w   = (const float*)d_in[10];
    const float* aproj_b   = (const float*)d_in[11];
    const float* ln2_w     = (const float*)d_in[12];
    const float* ln2_b     = (const float*)d_in[13];
    const float* fc_w      = (const float*)d_in[14];
    const float* fc_b      = (const float*)d_in[15];
    const float* cproj_w   = (const float*)d_in[16];
    const float* cproj_b   = (const float*)d_in[17];
    const float* ln_post_w = (const float*)d_in[18];
    const float* ln_post_b = (const float*)d_in[19];
    const float* proj_w    = (const float*)d_in[20];
    const float* head_w    = (const float*)d_in[21];
    const float* head_b    = (const float*)d_in[22];
    float* out = (float*)d_out;

    char* ws = (char*)d_ws;
    size_t off = 0;
    auto alloc = [&](size_t bytes) -> void* {
        off = (off + 255) & ~(size_t)255;
        void* p = ws + off;
        off += bytes;
        return p;
    };

    // bf16 weight copies
    __bf16* wc_conv  = (__bf16*)alloc((size_t)D_ * D_ * 2);
    __bf16* wc_qkv   = (__bf16*)alloc((size_t)DEPTH_ * 3 * D_ * D_ * 2);
    __bf16* wc_aproj = (__bf16*)alloc((size_t)DEPTH_ * D_ * D_ * 2);
    __bf16* wc_fc    = (__bf16*)alloc((size_t)DEPTH_ * MLP_ * D_ * 2);
    __bf16* wc_cproj = (__bf16*)alloc((size_t)DEPTH_ * D_ * MLP_ * 2);
    __bf16* wc_out   = (__bf16*)alloc((size_t)PROJ_ * D_ * 2);
    __bf16* wc_head  = (__bf16*)alloc((size_t)NCLSP * PROJ_ * 2);
    // activations
    float*  xbuf = (float*)alloc((size_t)NT_ * D_ * 4);         // residual stream
    __bf16* act  = (__bf16*)alloc((size_t)NT_ * MLP_ * 2);      // bf16 activations
    float*  tmpf = (float*)alloc((size_t)NT_ * MLP_ * 4);       // f32 GEMM out
    __bf16* qh   = (__bf16*)alloc((size_t)BH_ * NPAD * 64 * 2);
    __bf16* kh   = (__bf16*)alloc((size_t)BH_ * KPAD * 64 * 2);
    __bf16* vt   = (__bf16*)alloc((size_t)BH_ * 64 * KPAD * 2);
    float*  sc   = (float*)alloc((size_t)BH_ * NPAD * KPAD * 4);
    __bf16* at   = (__bf16*)alloc((size_t)BH_ * NPAD * KPAD * 2);
    float*  oh   = (float*)alloc((size_t)BH_ * NPAD * 64 * 4);
    float*  zbuf = (float*)alloc((size_t)B_ * PROJ_ * 4);
    __bf16* zbf  = (__bf16*)alloc((size_t)B_ * PROJ_ * 2);
    __bf16* clsbf= (__bf16*)alloc((size_t)B_ * D_ * 2);
    (void)ws_size;

    auto cvt = [&](const float* s, __bf16* d, long long n) {
        cvt_f32_bf16_kernel<<<cdiv_i(n, 256), 256, 0, stream>>>(s, d, n);
    };
    auto gemm = [&](const __bf16* A, int lda, long long sA,
                    const __bf16* W, int ldw, long long sW,
                    const float* bias, const float* add,
                    float* C, int ldc, long long sC,
                    int M, int Nfull, int Nreal, int K, float alpha, int batch) {
        dim3 grid(cdiv_i(Nfull, 64), cdiv_i(M, 64), batch);
        gemm_bf16_kernel<<<grid, 128, 0, stream>>>(A, lda, sA, W, ldw, sW, bias,
                                                   add, C, ldc, sC, M, Nfull,
                                                   Nreal, K, alpha);
    };

    // ---- weight conversion ----
    cvt(conv_w,  wc_conv,  (long long)D_ * D_);
    cvt(qkv_w,   wc_qkv,   (long long)DEPTH_ * 3 * D_ * D_);
    cvt(aproj_w, wc_aproj, (long long)DEPTH_ * D_ * D_);
    cvt(fc_w,    wc_fc,    (long long)DEPTH_ * MLP_ * D_);
    cvt(cproj_w, wc_cproj, (long long)DEPTH_ * D_ * MLP_);
    cvt(proj_w,  wc_out,   (long long)PROJ_ * D_);
    {
        long long n = (long long)NCLSP * PROJ_;
        cvt_pad_rows_kernel<<<cdiv_i(n, 256), 256, 0, stream>>>(head_w, wc_head,
                                                                NCLS, PROJ_, n);
    }

    // ---- patch embedding ----
    {
        long long n = (long long)MPATCH * D_;
        im2col_kernel<<<cdiv_i(n, 256), 256, 0, stream>>>(x_in, act, n);
    }
    gemm(act, D_, 0, wc_conv, D_, 0, nullptr, nullptr,
         tmpf, D_, 0, MPATCH, D_, D_, D_, 1.0f, 1);
    {
        long long n = (long long)NT_ * D_;
        assemble_kernel<<<cdiv_i(n, 256), 256, 0, stream>>>(tmpf, cls_token,
                                                            pos_embed, xbuf, n);
    }
    ln_kernel<<<NT_, 256, 0, stream>>>(xbuf, D_, D_, ln_pre_w, ln_pre_b,
                                       xbuf, nullptr);

    // ---- transformer blocks ----
    for (int L = 0; L < DEPTH_; ++L) {
        const float* l1w = ln1_w + (long long)L * D_;
        const float* l1b = ln1_b + (long long)L * D_;
        const float* l2w = ln2_w + (long long)L * D_;
        const float* l2b = ln2_b + (long long)L * D_;
        const __bf16* Wqkv = wc_qkv + (long long)L * 3 * D_ * D_;
        const float*  bqkv = qkv_b + (long long)L * 3 * D_;
        const __bf16* Wap  = wc_aproj + (long long)L * D_ * D_;
        const float*  bap  = aproj_b + (long long)L * D_;
        const __bf16* Wfc  = wc_fc + (long long)L * MLP_ * D_;
        const float*  bfc  = fc_b + (long long)L * MLP_;
        const __bf16* Wcp  = wc_cproj + (long long)L * D_ * MLP_;
        const float*  bcp  = cproj_b + (long long)L * D_;

        // h = LN1(x) -> bf16
        ln_kernel<<<NT_, 256, 0, stream>>>(xbuf, D_, D_, l1w, l1b, nullptr, act);
        // qkv = h @ Wqkv^T + b  -> tmpf [6304, 2304]
        gemm(act, D_, 0, Wqkv, D_, 0, bqkv, nullptr,
             tmpf, 3 * D_, 0, NT_, 3 * D_, 3 * D_, D_, 1.0f, 1);
        // scatter to padded head layouts
        {
            long long nq = (long long)BH_ * NPAD * 64;
            fill_qk_kernel<<<cdiv_i(nq, 256), 256, 0, stream>>>(tmpf, qh, NPAD, 0, nq);
            long long nk = (long long)BH_ * KPAD * 64;
            fill_qk_kernel<<<cdiv_i(nk, 256), 256, 0, stream>>>(tmpf, kh, KPAD, 1, nk);
            long long nv = (long long)BH_ * 64 * KPAD;
            fill_vt_kernel<<<cdiv_i(nv, 256), 256, 0, stream>>>(tmpf, vt, nv);
        }
        // scores = (Q @ K^T) * 1/8  -> sc [BH, 208, 224]
        gemm(qh, 64, (long long)NPAD * 64,
             kh, 64, (long long)KPAD * 64,
             nullptr, nullptr,
             sc, KPAD, (long long)NPAD * KPAD,
             NPAD, KPAD, KPAD, DH_, 0.125f, BH_);
        // softmax (masked to 197) -> at bf16
        softmax_kernel<<<BH_ * NPAD, 256, 0, stream>>>(sc, at);
        // O = attn @ V  -> oh [BH, 208, 64]   (W = V^T, [64, 224])
        gemm(at, KPAD, (long long)NPAD * KPAD,
             vt, KPAD, (long long)64 * KPAD,
             nullptr, nullptr,
             oh, 64, (long long)NPAD * 64,
             NPAD, 64, 64, KPAD, 1.0f, BH_);
        // gather heads back to token stream bf16
        {
            long long n = (long long)NT_ * D_;
            gather_o_kernel<<<cdiv_i(n, 256), 256, 0, stream>>>(oh, act, n);
        }
        // x = x + o @ Wap^T + bap   (fused residual)
        gemm(act, D_, 0, Wap, D_, 0, bap, xbuf,
             xbuf, D_, 0, NT_, D_, D_, D_, 1.0f, 1);
        // h = LN2(x) -> bf16
        ln_kernel<<<NT_, 256, 0, stream>>>(xbuf, D_, D_, l2w, l2b, nullptr, act);
        // fc = h @ Wfc^T + bfc -> tmpf [6304, 3072]
        gemm(act, D_, 0, Wfc, D_, 0, bfc, nullptr,
             tmpf, MLP_, 0, NT_, MLP_, MLP_, D_, 1.0f, 1);
        // GELU -> bf16
        {
            long long n = (long long)NT_ * MLP_;
            gelu_kernel<<<cdiv_i(n, 256), 256, 0, stream>>>(tmpf, act, n);
        }
        // x = x + gelu @ Wcp^T + bcp  (fused residual)
        gemm(act, MLP_, 0, Wcp, MLP_, 0, bcp, xbuf,
             xbuf, D_, 0, NT_, D_, D_, MLP_, 1.0f, 1);
    }

    // ---- head ----
    // ln_post on the 32 cls rows (row stride 197*768)
    ln_kernel<<<B_, 256, 0, stream>>>(xbuf, (long long)NTOK * D_, D_,
                                      ln_post_w, ln_post_b, nullptr, clsbf);
    // z = cls @ proj_w^T  -> zbuf [32, 512]
    gemm(clsbf, D_, 0, wc_out, D_, 0, nullptr, nullptr,
         zbuf, PROJ_, 0, B_, PROJ_, PROJ_, D_, 1.0f, 1);
    cvt(zbuf, zbf, (long long)B_ * PROJ_);
    // out = z @ head_w^T + head_b  (Nreal = 1000, padded to 1008)
    gemm(zbf, PROJ_, 0, wc_head, PROJ_, 0, head_b, nullptr,
         out, NCLS, 0, B_, NCLSP, NCLS, PROJ_, 1.0f, 1);
    (void)in_sizes; (void)n_in; (void)out_size;
}
